// MultipathDynamicAttention_39788577030342
// MI455X (gfx1250) — matrix operand order repaired and checked by reading, not compiled, not run on hardware
//
#include <hip/hip_runtime.h>
#include <hip/hip_bf16.h>

#define BB   4
#define CC   64
#define HH_  64
#define WW_  64
#define NN   4096
#define EE   64
#define TC   192
#define KTOP 128

typedef __attribute__((ext_vector_type(16))) _Float16 v16h;
typedef __attribute__((ext_vector_type(8)))  _Float16 v8h;
typedef __attribute__((ext_vector_type(8)))  float    v8f;

__device__ __forceinline__ v8f wmma16(v16h a, v16h b, v8f c) {
  return __builtin_amdgcn_wmma_f32_16x16x32_f16(false, a, false, b, (short)0, c, false, false);
}

// A-fragment: two contiguous 16-byte chunks (K = [base+16h, base+16h+8) and [base+16h+8, base+16h+16))
__device__ __forceinline__ v16h load_a_frag(const _Float16* row, int kbase, int hh) {
  v8h lo = *reinterpret_cast<const v8h*>(row + kbase + hh * 16);
  v8h hi = *reinterpret_cast<const v8h*>(row + kbase + hh * 16 + 8);
  return __builtin_shufflevector(lo, hi, 0, 1, 2, 3, 4, 5, 6, 7, 8, 9, 10, 11, 12, 13, 14, 15);
}
// B-fragment from 32 contiguous bytes
__device__ __forceinline__ v16h load_b_frag(const _Float16* p) {
  return *reinterpret_cast<const v16h*>(p);
}

// reductions across the 16 lanes of a half-wave (xor masks 1..8 stay inside the half)
__device__ __forceinline__ float red_max16(float v) {
#pragma unroll
  for (int off = 1; off < 16; off <<= 1) v = fmaxf(v, __shfl_xor(v, off, 32));
  return v;
}
__device__ __forceinline__ float red_sum16(float v) {
#pragma unroll
  for (int off = 1; off < 16; off <<= 1) v += __shfl_xor(v, off, 32);
  return v;
}

// ---------------- 1. fused multi-scale conv -> f_flat f16 [B,N,192] ----------------
__global__ void k_conv_fflat(const float* __restrict__ x,
                             const float* __restrict__ w1, const float* __restrict__ b1,
                             const float* __restrict__ w3, const float* __restrict__ b3,
                             const float* __restrict__ w5, const float* __restrict__ b5,
                             _Float16* __restrict__ fflat) {
  int t = blockIdx.x * blockDim.x + threadIdx.x;
  if (t >= BB * NN * TC) return;
  int j = t % TC;
  int n = (t / TC) % NN;
  int b = t / (TC * NN);
  int h = n >> 6, w = n & 63;
  const float* xb = x + (size_t)b * CC * NN;
  float acc;
  if (j < CC) {                       // 1x1
    int oc = j;
    acc = b1[oc];
    const float* wr = w1 + oc * CC;
    for (int ic = 0; ic < CC; ++ic) acc += xb[ic * NN + n] * wr[ic];
  } else if (j < 2 * CC) {            // 3x3 pad1
    int oc = j - CC;
    acc = b3[oc];
    for (int ic = 0; ic < CC; ++ic) {
      const float* xc = xb + ic * NN;
      const float* wr = w3 + (oc * CC + ic) * 9;
      for (int ky = 0; ky < 3; ++ky) {
        int hy = h + ky - 1;
        if (hy < 0 || hy >= HH_) continue;
        for (int kx = 0; kx < 3; ++kx) {
          int wx = w + kx - 1;
          if (wx < 0 || wx >= WW_) continue;
          acc += xc[hy * WW_ + wx] * wr[ky * 3 + kx];
        }
      }
    }
  } else {                            // 5x5 pad2
    int oc = j - 2 * CC;
    acc = b5[oc];
    for (int ic = 0; ic < CC; ++ic) {
      const float* xc = xb + ic * NN;
      const float* wr = w5 + (oc * CC + ic) * 25;
      for (int ky = 0; ky < 5; ++ky) {
        int hy = h + ky - 2;
        if (hy < 0 || hy >= HH_) continue;
        for (int kx = 0; kx < 5; ++kx) {
          int wx = w + kx - 2;
          if (wx < 0 || wx >= WW_) continue;
          acc += xc[hy * WW_ + wx] * wr[ky * 5 + kx];
        }
      }
    }
  }
  fflat[(size_t)(b * NN + n) * TC + j] = (_Float16)acc;
}

// ---------------- 2. weight prep: f32 -> f16, B-fragment-native tiled layout ----------------
// Wt[((kc*2+h)*TC + n)*16 + j] = Wcat[(kc*32 + h*16 + j)][n]   (kc<6)
// Wot[((kc*2+h)*CC + n)*16 + j] = wo[(kc*32 + h*16 + j)][n]    (kc<2)
__global__ void k_prep_weights(const float* __restrict__ wq, const float* __restrict__ wk,
                               const float* __restrict__ wv, const float* __restrict__ wo,
                               _Float16* __restrict__ Wt, _Float16* __restrict__ Wot) {
  int t = blockIdx.x * blockDim.x + threadIdx.x;
  if (t < TC * TC) {
    int j = t & 15;
    int n = (t >> 4) % TC;
    int rest = t / (16 * TC);
    int krow = (rest >> 1) * 32 + (rest & 1) * 16 + j;   // 0..191
    float v = (n < CC) ? wq[krow * EE + n]
            : (n < 2 * CC) ? wk[krow * EE + (n - CC)]
            : wv[krow * CC + (n - 2 * CC)];
    Wt[t] = (_Float16)v;
  } else if (t < TC * TC + CC * CC) {
    int t2 = t - TC * TC;
    int j = t2 & 15;
    int n = (t2 >> 4) % CC;
    int rest = t2 / (16 * CC);
    int krow = (rest >> 1) * 32 + (rest & 1) * 16 + j;   // 0..63
    Wot[t2] = (_Float16)wo[krow * CC + n];
  }
}

// ---------------- 3. QKV: [4096x192] @ [192x192] via WMMA ----------------
__global__ void k_qkv_gemm(const _Float16* __restrict__ fflat, const _Float16* __restrict__ Wt,
                           const float* __restrict__ bq, const float* __restrict__ bk,
                           const float* __restrict__ bv,
                           _Float16* __restrict__ Qh, _Float16* __restrict__ Kh,
                           _Float16* __restrict__ Vh) {
  int wave = (blockIdx.x * blockDim.x + threadIdx.x) >> 5;
  int lane = threadIdx.x & 31;
  int ct = wave % 12;
  int rt = (wave / 12) & 255;
  int b  = wave / (12 * 256);
  int mrow = lane & 15, hh = lane >> 4;
  const _Float16* Arow = fflat + (size_t)(b * NN + rt * 16 + mrow) * TC;
  int col = ct * 16 + mrow;
  v8f c = {};
#pragma unroll
  for (int kc = 0; kc < 6; ++kc) {
    v16h a  = load_a_frag(Arow, kc * 32, hh);
    v16h bf = load_b_frag(Wt + (size_t)((kc * 2 + hh) * TC + col) * 16);
    c = wmma16(a, bf, c);
  }
#pragma unroll
  for (int g = 0; g < 8; ++g) {
    int row = rt * 16 + g + 8 * hh;
    size_t base = (size_t)(b * NN + row) * EE;
    float v = c[g];
    if (col < CC)          Qh[base + col]            = (_Float16)(v + bq[col]);
    else if (col < 2 * CC) Kh[base + (col - CC)]     = (_Float16)(v + bk[col - CC]);
    else                   Vh[base + (col - 2 * CC)] = (_Float16)(v + bv[col - 2 * CC]);
  }
}

// ---------------- 4. pass A: flash-style row max / sumexp ----------------
__global__ void k_attn_stats(const _Float16* __restrict__ Qh, const _Float16* __restrict__ Kh,
                             float* __restrict__ rowmax, float* __restrict__ rowsum) {
  int wave = (blockIdx.x * blockDim.x + threadIdx.x) >> 5;
  int lane = threadIdx.x & 31;
  int qt = wave & 255;
  int b  = wave >> 8;
  int mrow = lane & 15, hh = lane >> 4;
  const _Float16* Arow = Qh + (size_t)(b * NN + qt * 16 + mrow) * EE;
  v16h a0 = load_a_frag(Arow, 0, hh);
  v16h a1 = load_a_frag(Arow, 32, hh);
  float mr[8], lr[8];
#pragma unroll
  for (int g = 0; g < 8; ++g) { mr[g] = -1e30f; lr[g] = 0.f; }
  for (int kt = 0; kt < 256; ++kt) {
    const _Float16* Kbase = Kh + (size_t)(b * NN + kt * 16 + mrow) * EE;
    if (kt < 255) __builtin_prefetch(Kbase + 16 * EE, 0, 1);
    v16h bf0 = load_b_frag(Kbase + hh * 16);
    v16h bf1 = load_b_frag(Kbase + 32 + hh * 16);
    v8f c = {};
    c = wmma16(a0, bf0, c);
    c = wmma16(a1, bf1, c);
#pragma unroll
    for (int g = 0; g < 8; ++g) {
      float s   = c[g] * 0.125f;
      float rmx = red_max16(s);
      float nm  = fmaxf(mr[g], rmx);
      float p   = __expf(s - nm);
      float rs  = red_sum16(p);
      lr[g] = lr[g] * __expf(mr[g] - nm) + rs;
      mr[g] = nm;
    }
  }
  if (mrow == 0) {
#pragma unroll
    for (int g = 0; g < 8; ++g) {
      int row = qt * 16 + g + 8 * hh;
      rowmax[b * NN + row] = mr[g];
      rowsum[b * NN + row] = lr[g];
    }
  }
}

// ---------------- 5. pass B: key importance (column sums), S' = K.Q^T ----------------
__global__ void k_colsum(const _Float16* __restrict__ Qh, const _Float16* __restrict__ Kh,
                         const float* __restrict__ rowmax, const float* __restrict__ rowsum,
                         float* __restrict__ kimp) {
  int wave = (blockIdx.x * blockDim.x + threadIdx.x) >> 5;
  int lane = threadIdx.x & 31;
  int kt = wave & 255;
  int b  = wave >> 8;
  int mrow = lane & 15, hh = lane >> 4;
  const _Float16* Arow = Kh + (size_t)(b * NN + kt * 16 + mrow) * EE;
  v16h a0 = load_a_frag(Arow, 0, hh);
  v16h a1 = load_a_frag(Arow, 32, hh);
  float acc[8];
#pragma unroll
  for (int g = 0; g < 8; ++g) acc[g] = 0.f;
  for (int qt = 0; qt < 256; ++qt) {
    int q = qt * 16 + mrow;
    float mq   = rowmax[b * NN + q];
    float linv = 1.0f / rowsum[b * NN + q];
    const _Float16* Qbase = Qh + (size_t)(b * NN + q) * EE;
    if (qt < 255) __builtin_prefetch(Qbase + 16 * EE, 0, 1);
    v16h bf0 = load_b_frag(Qbase + hh * 16);
    v16h bf1 = load_b_frag(Qbase + 32 + hh * 16);
    v8f c = {};
    c = wmma16(a0, bf0, c);
    c = wmma16(a1, bf1, c);
#pragma unroll
    for (int g = 0; g < 8; ++g)
      acc[g] += __expf(c[g] * 0.125f - mq) * linv;
  }
#pragma unroll
  for (int g = 0; g < 8; ++g) {
    float s = red_sum16(acc[g]);
    if (mrow == 0) kimp[b * NN + kt * 16 + g + 8 * hh] = s;
  }
}

// ---------------- 6. top-k (128 of 4096) + gather V_sel in B-frag layout ----------------
// Vselt[b][((s>>4)*EE + n)*16 + (s&15)] = Vh[b][chosen[s]][n]
__global__ void k_topk_gather(const float* __restrict__ kimp, const _Float16* __restrict__ Vh,
                              int* __restrict__ topk, _Float16* __restrict__ Vselt) {
  __shared__ float vals[NN];
  __shared__ float smax[256];
  __shared__ int   sidx[256];
  __shared__ int   chosen[KTOP];
  int b = blockIdx.x, t = threadIdx.x;
  for (int i = t; i < NN; i += 256) vals[i] = kimp[b * NN + i];
  __syncthreads();
  for (int it = 0; it < KTOP; ++it) {
    float bm = -1e30f; int bi = NN;
    for (int i = t; i < NN; i += 256) {
      float v = vals[i];
      if (v > bm || (v == bm && i < bi)) { bm = v; bi = i; }
    }
    smax[t] = bm; sidx[t] = bi;
    __syncthreads();
    for (int s = 128; s > 0; s >>= 1) {
      if (t < s) {
        if (smax[t + s] > smax[t] || (smax[t + s] == smax[t] && sidx[t + s] < sidx[t])) {
          smax[t] = smax[t + s]; sidx[t] = sidx[t + s];
        }
      }
      __syncthreads();
    }
    if (t == 0) {
      chosen[it] = sidx[0];
      topk[b * KTOP + it] = sidx[0];
      vals[sidx[0]] = -1e30f;
    }
    __syncthreads();
  }
  for (int i = t; i < KTOP * EE; i += 256) {
    int s = i >> 6, n = i & 63;
    Vselt[(size_t)b * KTOP * EE + (size_t)((s >> 4) * EE + n) * 16 + (s & 15)] =
        Vh[((size_t)b * NN + chosen[s]) * EE + n];
  }
}

// ---------------- 7. selected softmax probs P_sel f16 [B,N,128] ----------------
__global__ void k_psel(const _Float16* __restrict__ Qh, const _Float16* __restrict__ Kh,
                       const int* __restrict__ topk,
                       const float* __restrict__ rowmax, const float* __restrict__ rowsum,
                       _Float16* __restrict__ Psel) {
  int wave = (blockIdx.x * blockDim.x + threadIdx.x) >> 5;
  int lane = threadIdx.x & 31;
  int qt = wave & 255;
  int b  = wave >> 8;
  int mrow = lane & 15, hh = lane >> 4;
  const _Float16* Arow = Qh + (size_t)(b * NN + qt * 16 + mrow) * EE;
  v16h a0 = load_a_frag(Arow, 0, hh);
  v16h a1 = load_a_frag(Arow, 32, hh);
  float mr[8], li[8];
#pragma unroll
  for (int g = 0; g < 8; ++g) {
    int row = qt * 16 + g + 8 * hh;
    mr[g] = rowmax[b * NN + row];
    li[g] = 1.0f / rowsum[b * NN + row];
  }
  for (int st = 0; st < 8; ++st) {
    int kk   = st * 16 + mrow;
    int kidx = topk[b * KTOP + kk];
    const _Float16* Kbase = Kh + ((size_t)b * NN + kidx) * EE;
    v16h bf0 = load_b_frag(Kbase + hh * 16);
    v16h bf1 = load_b_frag(Kbase + 32 + hh * 16);
    v8f c = {};
    c = wmma16(a0, bf0, c);
    c = wmma16(a1, bf1, c);
#pragma unroll
    for (int g = 0; g < 8; ++g) {
      float p = __expf(c[g] * 0.125f - mr[g]) * li[g];
      int row = qt * 16 + g + 8 * hh;
      Psel[(size_t)(b * NN + row) * KTOP + kk] = (_Float16)p;
    }
  }
}

// ---------------- 8. AV: [4096x128] @ [128x64] ----------------
__global__ void k_av_gemm(const _Float16* __restrict__ Psel, const _Float16* __restrict__ Vselt,
                          _Float16* __restrict__ Oatt) {
  int wave = (blockIdx.x * blockDim.x + threadIdx.x) >> 5;
  int lane = threadIdx.x & 31;
  int qt = wave & 255;
  int b  = wave >> 8;
  int mrow = lane & 15, hh = lane >> 4;
  const _Float16* Arow = Psel + (size_t)(b * NN + qt * 16 + mrow) * KTOP;
  v16h a[4];
#pragma unroll
  for (int kc = 0; kc < 4; ++kc) a[kc] = load_a_frag(Arow, kc * 32, hh);
  const _Float16* Vb = Vselt + (size_t)b * KTOP * EE;
#pragma unroll
  for (int ctl = 0; ctl < 4; ++ctl) {
    v8f c = {};
#pragma unroll
    for (int kc = 0; kc < 4; ++kc) {
      v16h bf = load_b_frag(Vb + (size_t)((kc * 2 + hh) * EE + ctl * 16 + mrow) * 16);
      c = wmma16(a[kc], bf, c);
    }
#pragma unroll
    for (int g = 0; g < 8; ++g) {
      int row = qt * 16 + g + 8 * hh;
      Oatt[(size_t)(b * NN + row) * EE + ctl * 16 + mrow] = (_Float16)c[g];
    }
  }
}

// ---------------- 9. output proj (+bo), write NCHW f32 ----------------
__global__ void k_out_proj(const _Float16* __restrict__ Oatt, const _Float16* __restrict__ Wot,
                           const float* __restrict__ bo, float* __restrict__ out) {
  int wave = (blockIdx.x * blockDim.x + threadIdx.x) >> 5;
  int lane = threadIdx.x & 31;
  int qt = wave & 255;
  int b  = wave >> 8;
  int mrow = lane & 15, hh = lane >> 4;
  const _Float16* Arow = Oatt + (size_t)(b * NN + qt * 16 + mrow) * EE;
  v16h a0 = load_a_frag(Arow, 0, hh);
  v16h a1 = load_a_frag(Arow, 32, hh);
#pragma unroll
  for (int ctl = 0; ctl < 4; ++ctl) {
    int col = ctl * 16 + mrow;
    v16h bf0 = load_b_frag(Wot + (size_t)((0 * 2 + hh) * CC + col) * 16);
    v16h bf1 = load_b_frag(Wot + (size_t)((1 * 2 + hh) * CC + col) * 16);
    v8f c = {};
    c = wmma16(a0, bf0, c);
    c = wmma16(a1, bf1, c);
    float bias = bo[col];
#pragma unroll
    for (int g = 0; g < 8; ++g) {
      int row = qt * 16 + g + 8 * hh;
      out[(size_t)(b * CC + col) * NN + row] = c[g] + bias;
    }
  }
}

extern "C" void kernel_launch(void* const* d_in, const int* in_sizes, int n_in,
                              void* d_out, int out_size, void* d_ws, size_t ws_size,
                              hipStream_t stream) {
  (void)in_sizes; (void)n_in; (void)out_size; (void)ws_size;
  const float* x  = (const float*)d_in[0];
  const float* w1 = (const float*)d_in[1];
  const float* b1 = (const float*)d_in[2];
  const float* w3 = (const float*)d_in[3];
  const float* b3 = (const float*)d_in[4];
  const float* w5 = (const float*)d_in[5];
  const float* b5 = (const float*)d_in[6];
  const float* wq = (const float*)d_in[7];
  const float* bq = (const float*)d_in[8];
  const float* wk = (const float*)d_in[9];
  const float* bk = (const float*)d_in[10];
  const float* wv = (const float*)d_in[11];
  const float* bv = (const float*)d_in[12];
  const float* wo = (const float*)d_in[13];
  const float* bo = (const float*)d_in[14];
  float* out = (float*)d_out;

  char* ws = (char*)d_ws;
  size_t off = 0;
  auto carve = [&](size_t bytes) -> void* {
    void* p = ws + off;
    off += (bytes + 255) & ~(size_t)255;
    return p;
  };
  _Float16* fflat = (_Float16*)carve((size_t)BB * NN * TC * 2);
  _Float16* Wt    = (_Float16*)carve((size_t)TC * TC * 2);
  _Float16* Wot   = (_Float16*)carve((size_t)CC * CC * 2);
  _Float16* Qh    = (_Float16*)carve((size_t)BB * NN * EE * 2);
  _Float16* Kh    = (_Float16*)carve((size_t)BB * NN * EE * 2);
  _Float16* Vh    = (_Float16*)carve((size_t)BB * NN * EE * 2);
  float*    rmax  = (float*)carve((size_t)BB * NN * 4);
  float*    rsum  = (float*)carve((size_t)BB * NN * 4);
  float*    kimp  = (float*)carve((size_t)BB * NN * 4);
  int*      topk  = (int*)carve((size_t)BB * KTOP * 4);
  _Float16* Vselt = (_Float16*)carve((size_t)BB * KTOP * EE * 2);
  _Float16* Psel  = (_Float16*)carve((size_t)BB * NN * KTOP * 2);
  _Float16* Oatt  = (_Float16*)carve((size_t)BB * NN * EE * 2);

  dim3 blk(256);
  k_conv_fflat<<<(BB * NN * TC + 255) / 256, blk, 0, stream>>>(x, w1, b1, w3, b3, w5, b5, fflat);
  k_prep_weights<<<(TC * TC + CC * CC + 255) / 256, blk, 0, stream>>>(wq, wk, wv, wo, Wt, Wot);
  k_qkv_gemm<<<(BB * 256 * 12) / 8, blk, 0, stream>>>(fflat, Wt, bq, bk, bv, Qh, Kh, Vh);
  k_attn_stats<<<(BB * 256) / 8, blk, 0, stream>>>(Qh, Kh, rmax, rsum);
  k_colsum<<<(BB * 256) / 8, blk, 0, stream>>>(Qh, Kh, rmax, rsum, kimp);
  k_topk_gather<<<BB, blk, 0, stream>>>(kimp, Vh, topk, Vselt);
  k_psel<<<(BB * 256) / 8, blk, 0, stream>>>(Qh, Kh, topk, rmax, rsum, Psel);
  k_av_gemm<<<(BB * 256) / 8, blk, 0, stream>>>(Psel, Vselt, Oatt);
  k_out_proj<<<(BB * 256) / 8, blk, 0, stream>>>(Oatt, Wot, bo, out);
}
